// KPlaneField_84963043049523
// MI455X (gfx1250) — compile-verified, matrix operand order checked by reading, not run on hardware
//
#include <hip/hip_runtime.h>

// ---------------------------------------------------------------------------
// K-Planes field (featurize + CondMLP) for MI455X / gfx1250, wave32 + WMMA.
// GEMMs use v_wmma_f32_16x16x32_bf16 (bf16 in, f32 accumulate).
// bf16<->f32 conversions use native casts so the compiler emits v_cvt_*bf16*
// instead of manual integer rounding chains.
// ---------------------------------------------------------------------------

typedef __attribute__((ext_vector_type(16))) __bf16 v16bf;
typedef __attribute__((ext_vector_type(8)))  __bf16 v8bf;
typedef __attribute__((ext_vector_type(8)))  float  v8f;

#define NROWS_BLK 128      // points per workgroup (4 waves x 32 rows)
#define STRIDE    264      // LDS row stride in bf16 elems (256 + 8 pad, 16B aligned)

static __device__ __forceinline__ __bf16 f2bf(float f) {
  return (__bf16)f;                       // native fptrunc -> v_cvt bf16
}
static __device__ __forceinline__ float bf2f(__bf16 b) {
  return (float)b;                        // native fpext
}

// ---------------------------------------------------------------------------
// Weight packing: fp32 [K][N] row-major -> bf16 WMMA B-fragments.
// Fragment tile = 32(K) x 16(N), 512 bf16 = 1KB, stored [nt][kt] per layer.
// Per-lane layout (CDNA5 ISA 7.12.2, 16-bit B 32x16):
//   col = nt*16 + (lane&15); kbase = kt*32 + (lane>=16 ? 8 : 0)
//   elems 0..7 -> K = kbase+e ; elems 8..15 -> K = kbase+16+(e-8)
// Tiles: l0_w 96x256 -> 3*16 = 48 ; res_w 9x(256x256) -> 9*128 ; l1_w 256x16 -> 8
// ---------------------------------------------------------------------------
__global__ __launch_bounds__(32)
void pack_weights(const float* __restrict__ l0_w,
                  const float* __restrict__ res_w,
                  const float* __restrict__ l1_w,
                  __bf16* __restrict__ out) {
  const int tile = blockIdx.x;       // 0..1207
  const int lane = threadIdx.x;      // 0..31
  const float* W; int ktiles, fout, t;
  if (tile < 48)        { W = l0_w;  ktiles = 3; fout = 256; t = tile; }
  else if (tile < 1200) { int r = tile - 48;
                          W = res_w + (size_t)(r >> 7) * 65536;
                          ktiles = 8; fout = 256; t = r & 127; }
  else                  { W = l1_w;  ktiles = 8; fout = 16;  t = tile - 1200; }
  const int nt = t / ktiles, kt = t % ktiles;
  __bf16* dst = out + ((size_t)tile << 9) + (lane << 4);
  const int col = (nt << 4) + (lane & 15);
  const int kb  = (kt << 5) + ((lane >> 4) << 3);
#pragma unroll
  for (int e = 0; e < 16; ++e) {
    int k = kb + (e < 8 ? e : e + 8);           // e>=8 -> kbase+16+(e-8)
    dst[e] = f2bf(W[(size_t)k * fout + col]);
  }
}

// ---------------------------------------------------------------------------
// A-fragment load from LDS activations (bf16 row-major, stride STRIDE).
// CDNA5 16-bit A 16x32 layout: row = lane&15, kbase = (lane>=16?8:0);
// elems 0..7 = K kbase..kbase+7, elems 8..15 = K kbase+16..kbase+23.
// -> two contiguous 16B chunks (ds_load_b128 pair).
// ---------------------------------------------------------------------------
static __device__ __forceinline__ v16bf load_a(const __bf16* buf, int lane,
                                               int row_base, int kt) {
  const int row = row_base + (lane & 15);
  const int kb  = (kt << 5) + ((lane >> 4) << 3);
  const __bf16* p = buf + row * STRIDE + kb;
  v8bf lo = *(const v8bf*)(p);
  v8bf hi = *(const v8bf*)(p + 16);
  v16bf a;
#pragma unroll
  for (int i = 0; i < 8; ++i) { a[i] = lo[i]; a[i + 8] = hi[i]; }
  return a;
}

enum { EP_L0 = 0, EP_Z = 1, EP_RELU = 2, EP_ADDRES = 3, EP_FINAL = 4 };

// One MLP layer for this wave's 32-row band: D[32 x NOUT] = A[32 x K] * W + b.
// Two 16-row M tiles per wave share every B fragment.
template<int EP, int K, int NOUT>
static __device__ __forceinline__ void gemm_layer(
    const __bf16* in, __bf16* out, const __bf16* res,
    const __bf16* __restrict__ w, const float* __restrict__ bias,
    int lane, int mwave, float* gout, int grow) {
  constexpr int KT = K / 32;
  constexpr int NT = NOUT / 16;
  const int roff = (lane >> 4) << 3;   // D rows: lanes 16..31 hold M+8
  const int cl   = lane & 15;

  for (int nt = 0; nt < NT; ++nt) {
    if (nt + 1 < NT)   // pull next column-tile of weights toward L2/L0
      __builtin_prefetch(w + (((size_t)(nt + 1) * KT) << 9) + (lane << 4), 0, 0);
    v8f acc0 = {0.f, 0.f, 0.f, 0.f, 0.f, 0.f, 0.f, 0.f};
    v8f acc1 = acc0;
#pragma unroll
    for (int kt = 0; kt < KT; ++kt) {
      v16bf a0 = load_a(in, lane, mwave,      kt);
      v16bf a1 = load_a(in, lane, mwave + 16, kt);
      v16bf bf = *(const v16bf*)(w + (((size_t)nt * KT + kt) << 9) + (lane << 4));
      acc0 = __builtin_amdgcn_wmma_f32_16x16x32_bf16(
                 false, a0, false, bf, (short)0, acc0, false, false);
      acc1 = __builtin_amdgcn_wmma_f32_16x16x32_bf16(
                 false, a1, false, bf, (short)0, acc1, false, false);
    }
    const int col = (nt << 4) + cl;
    const float bv = bias[col];
#pragma unroll
    for (int v = 0; v < 8; ++v) {
      const int r0 = mwave + roff + v;
      const int r1 = r0 + 16;
      float y0 = acc0[v] + bv;
      float y1 = acc1[v] + bv;
      if constexpr (EP == EP_L0) {                    // x = f@W + b
        out[r0 * STRIDE + col] = f2bf(y0);
        out[r1 * STRIDE + col] = f2bf(y1);
      } else if constexpr (EP == EP_Z) {              // res = x + relu(x@W+b)
        out[r0 * STRIDE + col] = f2bf(fmaxf(y0, 0.f) + bf2f(in[r0 * STRIDE + col]));
        out[r1 * STRIDE + col] = f2bf(fmaxf(y1, 0.f) + bf2f(in[r1 * STRIDE + col]));
      } else if constexpr (EP == EP_RELU) {           // t = relu(res@W+b)
        out[r0 * STRIDE + col] = f2bf(fmaxf(y0, 0.f));
        out[r1 * STRIDE + col] = f2bf(fmaxf(y1, 0.f));
      } else if constexpr (EP == EP_ADDRES) {         // x = relu(t@W+b) + res
        float v0 = fmaxf(y0, 0.f) + bf2f(res[r0 * STRIDE + col]);
        float v1 = fmaxf(y1, 0.f) + bf2f(res[r1 * STRIDE + col]);
        out[r0 * STRIDE + col] = f2bf(v0);            // element-wise RaW on res: safe
        out[r1 * STRIDE + col] = f2bf(v1);
      } else {                                        // EP_FINAL: sigmoid -> global
        gout[(size_t)(grow + r0) * 16 + col] = 1.0f / (1.0f + __expf(-y0));
        gout[(size_t)(grow + r1) * 16 + col] = 1.0f / (1.0f + __expf(-y1));
      }
    }
  }
}

struct Planes { const float* p[9]; };

__global__ __launch_bounds__(NROWS_BLK)
void kplane_kernel(const float* __restrict__ pts, const float* __restrict__ aabb,
                   Planes pl, const __bf16* __restrict__ wpack,
                   const float* __restrict__ l0_b, const float* __restrict__ res_b,
                   const float* __restrict__ l1_b, float* __restrict__ out) {
  __shared__ __bf16 sA[NROWS_BLK * STRIDE];   // 66 KB
  __shared__ __bf16 sB[NROWS_BLK * STRIDE];   // 66 KB  (132 KB -> 2 blocks/WGP)

  const int tid   = threadIdx.x;
  const int lane  = tid & 31;
  const int mwave = tid & ~31;                // wave's row band base (0/32/64/96)
  const int gbase = blockIdx.x * NROWS_BLK;

  // ---- featurize: 1 point per thread -> bf16 feats [row][0..95] in sA ----
  {
    const int p = gbase + tid;
    float xn[3];
#pragma unroll
    for (int d = 0; d < 3; ++d) {
      const float lo = aabb[d], hi = aabb[3 + d];
      xn[d] = (pts[p * 3 + d] - lo) * (2.0f / (hi - lo)) - 1.0f;
    }
    const int comb0[3] = {0, 0, 1};
    const int comb1[3] = {1, 2, 2};
    for (int s = 0; s < 3; ++s) {
      const int r = 128 << s;                 // SCALES 1,2,4 -> 128,256,512
      float acc[32];
#pragma unroll
      for (int c = 0; c < 32; ++c) acc[c] = 1.0f;
      for (int q = 0; q < 3; ++q) {
        const float* plane = pl.p[s * 3 + q];
        const float fx = (xn[comb0[q]] + 1.0f) * 0.5f * (float)(r - 1);
        const float fy = (xn[comb1[q]] + 1.0f) * 0.5f * (float)(r - 1);
        int x0 = (int)floorf(fx); x0 = x0 < 0 ? 0 : (x0 > r - 2 ? r - 2 : x0);
        int y0 = (int)floorf(fy); y0 = y0 < 0 ? 0 : (y0 > r - 2 ? r - 2 : y0);
        const float wx = fx - (float)x0, wy = fy - (float)y0;
        const float w00 = (1.f - wx) * (1.f - wy), w10 = wx * (1.f - wy);
        const float w01 = (1.f - wx) * wy,         w11 = wx * wy;
        const float* base = plane + (size_t)y0 * r + x0;
        const size_t rr = (size_t)r * r;
        for (int c = 0; c < 32; ++c) {
          const float* qp = base + c * rr;
          acc[c] *= qp[0] * w00 + qp[1] * w10 + qp[r] * w01 + qp[r + 1] * w11;
        }
      }
#pragma unroll
      for (int c = 0; c < 32; ++c)
        sA[tid * STRIDE + s * 32 + c] = f2bf(acc[c]);
    }
  }
  __syncthreads();

  // ---- CondMLP ----
  const __bf16* w = wpack;
  gemm_layer<EP_L0, 96, 256>(sA, sB, nullptr, w, l0_b, lane, mwave, nullptr, 0);
  w += 48 * 512;
  __syncthreads();

  __bf16* x = sB;       // current activations
  __bf16* o = sA;       // scratch / residual
  for (int i = 0; i < 3; ++i) {
    gemm_layer<EP_Z, 256, 256>(x, o, nullptr, w, res_b + (i * 3 + 0) * 256,
                               lane, mwave, nullptr, 0);
    w += 128 * 512; __syncthreads();
    gemm_layer<EP_RELU, 256, 256>(o, x, nullptr, w, res_b + (i * 3 + 1) * 256,
                                  lane, mwave, nullptr, 0);
    w += 128 * 512; __syncthreads();
    gemm_layer<EP_ADDRES, 256, 256>(x, o, o, w, res_b + (i * 3 + 2) * 256,
                                    lane, mwave, nullptr, 0);
    w += 128 * 512; __syncthreads();
    __bf16* t = x; x = o; o = t;              // new x ends in 'o'
  }
  gemm_layer<EP_FINAL, 256, 16>(x, nullptr, nullptr, w, l1_b, lane, mwave,
                                out, gbase);
}

// ---------------------------------------------------------------------------
// Host entry
// ---------------------------------------------------------------------------
extern "C" void kernel_launch(void* const* d_in, const int* in_sizes, int n_in,
                              void* d_out, int out_size, void* d_ws, size_t ws_size,
                              hipStream_t stream) {
  const float* pts   = (const float*)d_in[0];
  const float* aabb  = (const float*)d_in[1];
  Planes pl;
  for (int i = 0; i < 9; ++i) pl.p[i] = (const float*)d_in[2 + i];
  const float* l0_w  = (const float*)d_in[11];
  const float* l0_b  = (const float*)d_in[12];
  const float* res_w = (const float*)d_in[13];
  const float* res_b = (const float*)d_in[14];
  const float* l1_w  = (const float*)d_in[15];
  const float* l1_b  = (const float*)d_in[16];

  __bf16* wpack = (__bf16*)d_ws;              // 1208 tiles * 1KB = ~1.24 MB

  pack_weights<<<1208, 32, 0, stream>>>(l0_w, res_w, l1_w, wpack);

  const int n = in_sizes[0] / 3;              // 131072
  kplane_kernel<<<n / NROWS_BLK, NROWS_BLK, 0, stream>>>(
      pts, aabb, pl, wpack, l0_b, res_b, l1_b, (float*)d_out);
}
